// JacobiKANLinear_17411797418702
// MI455X (gfx1250) — matrix operand order repaired and checked by reading, not compiled, not run on hardware
//
#include <hip/hip_runtime.h>

// ---------------------------------------------------------------------------
// JacobiKAN linear:  out = silu(x) @ Wb^T + einsum('bik,oik', P(x), Jc) + bias
// Fused as ONE bf16 GEMM:  A' (8192 x 7168) * W'^T (1024 x 7168), f32 accum.
// GEMM is double-buffered in LDS with one barrier per K-step; tile staging
// uses gfx1250 GLOBAL_LOAD_ASYNC_TO_LDS_B128 (ASYNCcnt path) when available,
// else a register-staged software pipeline.
// ---------------------------------------------------------------------------

typedef __attribute__((ext_vector_type(16))) __bf16 bf16x16;
typedef __attribute__((ext_vector_type(8)))  __bf16 bf16x8;
typedef __attribute__((ext_vector_type(8)))  float  f32x8;
typedef __attribute__((__vector_size__(16))) int    i32x4;

#define BATCH   8192
#define IN_F    1024
#define OUT_F   1024
#define KTOT    7168            // 1024 + 1024*6
#define BM      128
#define BN      128
#define BK      32
#define LDSB    80              // LDS row: 64B data + 16B pad (16B aligned)
#define KSTEPS  (KTOT / BK)     // 224

#if defined(__HIP_DEVICE_COMPILE__) && \
    __has_builtin(__builtin_amdgcn_global_load_async_to_lds_b128)
#define USE_ASYNC 1
#else
#define USE_ASYNC 0
#endif

__device__ __forceinline__ unsigned short f2bf(float f) {
    union { float f; unsigned int u; } c; c.f = f;
    unsigned int u = c.u;
    u += 0x7fffu + ((u >> 16) & 1u);      // round-to-nearest-even
    return (unsigned short)(u >> 16);
}

#if USE_ASYNC
__device__ __forceinline__ void async_copy16(const void* g, void* l) {
    __builtin_amdgcn_global_load_async_to_lds_b128(
        (__attribute__((address_space(1))) i32x4*)g,
        (__attribute__((address_space(3))) i32x4*)l, 0, 0);
}
__device__ __forceinline__ void wait_async0() {
#if __has_builtin(__builtin_amdgcn_s_wait_asynccnt)
    __builtin_amdgcn_s_wait_asynccnt(0);
#else
    asm volatile("s_wait_asynccnt 0" ::: "memory");
#endif
}
#endif

// ---------------- prep: activations -> bf16 A' -----------------------------
__global__ __launch_bounds__(256) void prep_act(const float* __restrict__ x,
                                                unsigned short* __restrict__ Abf) {
    size_t idx = (size_t)blockIdx.x * 256 + threadIdx.x;   // over BATCH*IN_F
    if (idx >= (size_t)BATCH * IN_F) return;
    int b = (int)(idx >> 10);
    int i = (int)(idx & 1023);
    float v = x[idx];

    float sig = 1.0f / (1.0f + __expf(-v));
    Abf[(size_t)b * KTOT + i] = f2bf(v * sig);

    float t = tanhf(v);
    float p[6];
    p[0] = 1.0f;
    p[1] = 2.0f * t;                      // 0.5*(2*(A+1)*t + (A-B)), A=B=1
#pragma unroll
    for (int k = 2; k <= 5; ++k) {
        float kk    = (float)k;
        float alpha = 2.0f * kk * (kk + 2.0f) * (2.0f * kk);
        float gamma = (2.0f * kk) * (2.0f * kk + 1.0f) * (2.0f * kk + 2.0f);
        float delta = 2.0f * kk * kk * (2.0f * kk + 2.0f);
        p[k] = (alpha * t) / gamma * p[k - 1] - (delta / gamma) * p[k - 2];
    }
    unsigned short* dst = Abf + (size_t)b * KTOT + IN_F + (size_t)i * 6;
#pragma unroll
    for (int j = 0; j < 6; ++j) dst[j] = f2bf(p[j]);
}

// ---------------- prep: weights -> bf16 W' ---------------------------------
__global__ __launch_bounds__(256) void prep_w(const float* __restrict__ bw,
                                              const float* __restrict__ jc,
                                              unsigned short* __restrict__ Wbf) {
    size_t idx = (size_t)blockIdx.x * 256 + threadIdx.x;   // over OUT_F*KTOT
    if (idx >= (size_t)OUT_F * KTOT) return;
    size_t o = idx / KTOT;
    size_t k = idx % KTOT;
    float v = (k < IN_F) ? bw[o * IN_F + k]
                         : jc[o * (size_t)(IN_F * 6) + (k - IN_F)];
    Wbf[idx] = f2bf(v);
}

// ---------------- main GEMM: bf16 WMMA, f32 accumulate ---------------------
__global__ __launch_bounds__(256) void gemm_bf16(const unsigned short* __restrict__ Abf,
                                                 const unsigned short* __restrict__ Wbf,
                                                 const float* __restrict__ bias,
                                                 float* __restrict__ out) {
    __shared__ __attribute__((aligned(16))) unsigned char lsA[2][BM * LDSB];
    __shared__ __attribute__((aligned(16))) unsigned char lsB[2][BN * LDSB];

    const int tid     = threadIdx.x;
    const int rowBase = blockIdx.y * BM;      // batch
    const int colBase = blockIdx.x * BN;      // outputs

    const int w    = tid >> 5;                // wave 0..7
    const int lane = tid & 31;
    const int wm   = w & 3;                   // 32-row group within BM
    const int wn   = w >> 2;                  // 64-col group within BN
    const int half = lane >> 4;
    const int l16  = lane & 15;

    // staging map: thread -> rows r0 and r0+64, 16B chunk kc (BK*2 = 64B/row)
    const int r0 = tid >> 2;
    const int kc = tid & 3;
    const size_t rowOff2 = (size_t)64 * KTOT;                 // +64 rows (elements)
    const unsigned short* gA = Abf + (size_t)(rowBase + r0) * KTOT + kc * 8;
    const unsigned short* gW = Wbf + (size_t)(colBase + r0) * KTOT + kc * 8;
    const int lo  = r0 * LDSB + kc * 16;
    const int lo2 = (r0 + 64) * LDSB + kc * 16;

    f32x8 acc[2][4];
#pragma unroll
    for (int r = 0; r < 2; ++r)
#pragma unroll
        for (int c = 0; c < 4; ++c) acc[r][c] = (f32x8){0,0,0,0,0,0,0,0};

#if USE_ASYNC
    // prologue: tile 0 -> buffer 0 via async DMA, publish
    async_copy16(gA,           &lsA[0][lo]);
    async_copy16(gA + rowOff2, &lsA[0][lo2]);
    async_copy16(gW,           &lsB[0][lo]);
    async_copy16(gW + rowOff2, &lsB[0][lo2]);
    wait_async0();
    __syncthreads();
#else
    uint4 ra0 = *(const uint4*)(gA);
    uint4 ra1 = *(const uint4*)(gA + rowOff2);
    uint4 rw0 = *(const uint4*)(gW);
    uint4 rw1 = *(const uint4*)(gW + rowOff2);
#endif

    for (int kt = 0; kt < KSTEPS; ++kt) {
        const int cur = kt & 1;
        const int nxt = cur ^ 1;

#if USE_ASYNC
        // kick tile kt+1 into the other buffer; overlaps frag loads + WMMA
        if (kt + 1 < KSTEPS) {
            const unsigned short* gA1 = gA + (size_t)(kt + 1) * BK;
            const unsigned short* gW1 = gW + (size_t)(kt + 1) * BK;
            async_copy16(gA1,           &lsA[nxt][lo]);
            async_copy16(gA1 + rowOff2, &lsA[nxt][lo2]);
            async_copy16(gW1,           &lsB[nxt][lo]);
            async_copy16(gW1 + rowOff2, &lsB[nxt][lo2]);
        }
#else
        // publish tile kt (regs -> LDS), then prefetch tile kt+1 into regs
        *(uint4*)(&lsA[cur][lo])  = ra0;
        *(uint4*)(&lsA[cur][lo2]) = ra1;
        *(uint4*)(&lsB[cur][lo])  = rw0;
        *(uint4*)(&lsB[cur][lo2]) = rw1;
        __syncthreads();
        if (kt + 1 < KSTEPS) {
            const unsigned short* gA1 = gA + (size_t)(kt + 1) * BK;
            const unsigned short* gW1 = gW + (size_t)(kt + 1) * BK;
            ra0 = *(const uint4*)(gA1);
            ra1 = *(const uint4*)(gA1 + rowOff2);
            rw0 = *(const uint4*)(gW1);
            rw1 = *(const uint4*)(gW1 + rowOff2);
        }
#endif

        // ---- LDS -> fragments (ISA 16-bit A/B lane layouts) ---------------
        bf16x16 afrag[2], bfrag[4];
#pragma unroll
        for (int r = 0; r < 2; ++r) {
            const unsigned char* base = &lsA[cur][(wm * 32 + r * 16 + l16) * LDSB] + half * 16;
            ((bf16x8*)&afrag[r])[0] = *(const bf16x8*)(base);        // K 0..7  (+8 if half)
            ((bf16x8*)&afrag[r])[1] = *(const bf16x8*)(base + 32);   // K 16..23(+8 if half)
        }
#pragma unroll
        for (int c = 0; c < 4; ++c) {
            const unsigned char* base = &lsB[cur][(wn * 64 + c * 16 + l16) * LDSB] + half * 16;
            ((bf16x8*)&bfrag[c])[0] = *(const bf16x8*)(base);        // K 0..7  (+8 if half)
            ((bf16x8*)&bfrag[c])[1] = *(const bf16x8*)(base + 32);   // K 16..23(+8 if half)
        }

        // ---- 8 WMMAs: 32x64 C tile per wave -------------------------------
#pragma unroll
        for (int r = 0; r < 2; ++r)
#pragma unroll
            for (int c = 0; c < 4; ++c)
                acc[r][c] = __builtin_amdgcn_wmma_f32_16x16x32_bf16(
                    false, afrag[r], false, bfrag[c],
                    (short)0, acc[r][c], false, false);

#if USE_ASYNC
        if (kt + 1 < KSTEPS) wait_async0();   // my tile kt+1 writes landed
        __syncthreads();                      // everyone's landed; reads done
#endif
    }

    // ---- epilogue: C/D layout  M = v + 8*half, N = l16 --------------------
#pragma unroll
    for (int r = 0; r < 2; ++r)
#pragma unroll
        for (int c = 0; c < 4; ++c) {
            const int n = colBase + wn * 64 + c * 16 + l16;
            const float bv = bias[n];
#pragma unroll
            for (int v = 0; v < 8; ++v) {
                const int m = rowBase + wm * 32 + r * 16 + half * 8 + v;
                out[(size_t)m * OUT_F + n] = acc[r][c][v] + bv;
            }
        }
}

// ---------------------------------------------------------------------------
extern "C" void kernel_launch(void* const* d_in, const int* in_sizes, int n_in,
                              void* d_out, int out_size, void* d_ws, size_t ws_size,
                              hipStream_t stream) {
    const float* x    = (const float*)d_in[0];   // (8192, 1024)
    const float* bw   = (const float*)d_in[1];   // (1024, 1024)
    const float* jc   = (const float*)d_in[2];   // (1024, 1024, 6)
    const float* bias = (const float*)d_in[3];   // (1024,)
    float* out        = (float*)d_out;           // (8192, 1024)

    unsigned short* Abf = (unsigned short*)d_ws;                 // 8192*7168 bf16
    unsigned short* Wbf = Abf + (size_t)BATCH * KTOT;            // 1024*7168 bf16

    prep_act<<<(BATCH * IN_F) / 256, 256, 0, stream>>>(x, Abf);
    prep_w<<<(OUT_F * KTOT) / 256, 256, 0, stream>>>(bw, jc, Wbf);

    dim3 grid(OUT_F / BN, BATCH / BM);           // (8, 64)
    gemm_bf16<<<grid, 256, 0, stream>>>(Abf, Wbf, bias, out);
}
